// AttentionFusion_42099269435528
// MI455X (gfx1250) — matrix operand order repaired and checked by reading, compiled, not run on hardware
//
#include <hip/hip_runtime.h>
#include <hip/hip_bf16.h>
#include <stdint.h>

typedef __attribute__((ext_vector_type(2)))  float        v2f;
typedef __attribute__((ext_vector_type(8)))  float        v8f;
typedef __attribute__((ext_vector_type(8)))  unsigned int v8u;
typedef __attribute__((ext_vector_type(16))) __bf16       v16bf;

#define EMBED   1024
#define IN_DIM  2048
#define KTOT    4096        // concat(audio, visual) K dimension
#define KSTEP   32
#define LDSU    20          // LDS row stride in uints (80 bytes -> bank-conflict free)

// ---------------- bf16 helpers (round-to-nearest-even) ----------------
__device__ __forceinline__ unsigned bf16rnd(unsigned u) {
    return u + 0x7FFFu + ((u >> 16) & 1u);
}
__device__ __forceinline__ unsigned short f32_to_bf16(float f) {
    return (unsigned short)(bf16rnd(__float_as_uint(f)) >> 16);
}
__device__ __forceinline__ unsigned pack_bf16x2(float a, float b) {
    return (bf16rnd(__float_as_uint(a)) >> 16) | (bf16rnd(__float_as_uint(b)) & 0xFFFF0000u);
}

// =====================================================================
// f32 WMMA GEMM (NN, row-major): C[M,N] = A[M,K] @ B[K,N]
// One wave per 16x16 tile, V_WMMA_F32_16X16X4_F32, K stepped by 4.
// Used only for the small weight-composition GEMMs (f32 precision).
// =====================================================================
template <bool BF16OUT>
__global__ __launch_bounds__(32) void wgemm_f32(
    const float* __restrict__ A, int lda,
    const float* __restrict__ B, int ldb,
    void* __restrict__ C, int ldc, int K)
{
    const int lane = threadIdx.x;
    const int m    = lane & 15;
    const int hi   = lane >> 4;          // 0: K{0,1}, 1: K{2,3} per ISA 16x4 A layout
    const int m0   = blockIdx.y << 4;
    const int n0   = blockIdx.x << 4;

    v8f acc = {};
    const float* arow = A + (size_t)(m0 + m) * lda;
    for (int k = 0; k < K; k += 4) {
        v2f a, b;
        a.x = arow[k + 2 * hi];
        a.y = arow[k + 2 * hi + 1];
        b.x = B[(size_t)(k + 2 * hi)     * ldb + n0 + m];
        b.y = B[(size_t)(k + 2 * hi + 1) * ldb + n0 + m];
        acc = __builtin_amdgcn_wmma_f32_16x16x4_f32(false, a, false, b,
                                                    (short)0, acc, false, false);
    }
#pragma unroll
    for (int r = 0; r < 8; ++r) {
        const int row = m0 + r + 8 * hi;      // C layout: VGPR r -> M=r (+8 for hi lanes)
        const int col = n0 + m;
        if (BF16OUT)
            ((unsigned short*)C)[(size_t)row * ldc + col] = f32_to_bf16(acc[r]);
        else
            ((float*)C)[(size_t)row * ldc + col] = acc[r];
    }
}

// =====================================================================
// Bias folding (all reference biases are zeros, but computed generally)
// c    = Wo @ bvin + bo
// btot = G1@bv + G2@ba + (Wf1+Wf2)@c + bf
// =====================================================================
__global__ __launch_bounds__(256) void bias_c_kernel(
    const float* __restrict__ Wo, const float* __restrict__ bi,
    const float* __restrict__ bo, float* __restrict__ c)
{
    const int i = blockIdx.x * blockDim.x + threadIdx.x;
    if (i >= EMBED) return;
    const float* bvin = bi + 2 * EMBED;
    float s = bo[i];
    for (int j = 0; j < EMBED; ++j) s += Wo[(size_t)i * EMBED + j] * bvin[j];
    c[i] = s;
}

__global__ __launch_bounds__(256) void bias_tot_kernel(
    const float* __restrict__ Wf, const float* __restrict__ bf,
    const float* __restrict__ G1, const float* __restrict__ G2,
    const float* __restrict__ ba, const float* __restrict__ bv,
    const float* __restrict__ c,  float* __restrict__ btot)
{
    const int i = blockIdx.x * blockDim.x + threadIdx.x;
    if (i >= EMBED) return;
    float s = bf[i];
    for (int j = 0; j < EMBED; ++j) {
        s += (Wf[(size_t)i * 2 * EMBED + j] + Wf[(size_t)i * 2 * EMBED + EMBED + j]) * c[j];
        s += G1[(size_t)i * EMBED + j] * bv[j];
        s += G2[(size_t)i * EMBED + j] * ba[j];
    }
    btot[i] = s;
}

// =====================================================================
// Main fused GEMM: out = relu([audio|visual] @ Wcat^T + btot)
//   A: [32768, 4096] f32 (virtual concat), Wcat: [1024, 4096] bf16
// Block: 256 threads (8 waves, 4(M) x 2(N)), tile 128x128, K step 32.
// A staged via f32->bf16 convert into LDS; B staged via CDNA5 async
// global->LDS b128 copies (ASYNCcnt), double-buffered.
// =====================================================================
__global__ __launch_bounds__(256) void fused_out(
    const float* __restrict__ audio, const float* __restrict__ visual,
    const unsigned short* __restrict__ Wcat, const float* __restrict__ bias,
    float* __restrict__ out)
{
    __shared__ __align__(16) unsigned int ldsA[2][128 * LDSU];
    __shared__ __align__(16) unsigned int ldsB[2][128 * LDSU];

    const int tid  = threadIdx.x;
    const int lane = tid & 31;
    const int wv   = tid >> 5;
    const int wm   = wv >> 1;            // 0..3  (M wave coord)
    const int wn   = wv & 1;             // 0..1  (N wave coord)
    const int Mblk = blockIdx.y << 7;
    const int Nblk = blockIdx.x << 7;

    const int fr = tid >> 1;             // fill row 0..127
    const int fh = tid & 1;              // which 16-element K half

    const int lm    = lane & 15;
    const int ahalf = (lane < 16) ? 0 : 8;   // A K-chunk uint offset (kb=0 / kb=16)
    const int bhalf = (lane < 16) ? 0 : 8;   // B K-chunk uint offset (kb=0 / kb=16)
    const int mhi   = (lane < 16) ? 0 : 8;   // C/D row offset for hi lanes

    v8f acc[2][4] = {};

    auto fillA = [&](int buf, int k0) {
        const int kg = k0 + 16 * fh;
        const float* src = (kg < IN_DIM)
            ? (audio  + (size_t)(Mblk + fr) * IN_DIM + kg)
            : (visual + (size_t)(Mblk + fr) * IN_DIM + (kg - IN_DIM));
        const float4 f0 = ((const float4*)src)[0];
        const float4 f1 = ((const float4*)src)[1];
        const float4 f2 = ((const float4*)src)[2];
        const float4 f3 = ((const float4*)src)[3];
        uint4 p0, p1;
        p0.x = pack_bf16x2(f0.x, f0.y); p0.y = pack_bf16x2(f0.z, f0.w);
        p0.z = pack_bf16x2(f1.x, f1.y); p0.w = pack_bf16x2(f1.z, f1.w);
        p1.x = pack_bf16x2(f2.x, f2.y); p1.y = pack_bf16x2(f2.z, f2.w);
        p1.z = pack_bf16x2(f3.x, f3.y); p1.w = pack_bf16x2(f3.z, f3.w);
        unsigned int* d = &ldsA[buf][fr * LDSU + fh * 8];
        ((uint4*)d)[0]       = p0;
        ((uint4*)(d + 4))[0] = p1;
    };

    auto fillB = [&](int buf, int k0) {
        const unsigned long long g =
            (unsigned long long)(uintptr_t)(Wcat + (size_t)(Nblk + fr) * KTOT + k0 + 16 * fh);
        const unsigned loff =
            (unsigned)(uintptr_t)&ldsB[buf][fr * LDSU + fh * 8];
        asm volatile("global_load_async_to_lds_b128 %0, %1, off"
                     :: "v"(loff), "v"(g) : "memory");
        asm volatile("global_load_async_to_lds_b128 %0, %1, off offset:16"
                     :: "v"(loff), "v"(g) : "memory");
    };

    // prologue
    fillB(0, 0);
    fillA(0, 0);
    asm volatile("s_wait_asynccnt 0x0" ::: "memory");
    __syncthreads();

    int buf = 0;
    for (int kt = 0; kt < KTOT / KSTEP; ++kt) {
        const bool more = (kt + 1) < (KTOT / KSTEP);
        if (more) fillB(buf ^ 1, (kt + 1) * KSTEP);   // async copy overlaps compute

        v16bf aF[2], bF[4];
#pragma unroll
        for (int s = 0; s < 2; ++s) {
            const unsigned int* p = &ldsA[buf][(wm * 32 + s * 16 + lm) * LDSU + ahalf];
            const uint4 c0 = ((const uint4*)p)[0];        // K [kb, kb+8)
            const uint4 c1 = ((const uint4*)(p + 4))[0];  // K [kb+8, kb+16)
            v8u va = {c0.x, c0.y, c0.z, c0.w, c1.x, c1.y, c1.z, c1.w};
            aF[s] = __builtin_bit_cast(v16bf, va);
        }
#pragma unroll
        for (int t = 0; t < 4; ++t) {
            const unsigned int* p = &ldsB[buf][(wn * 64 + t * 16 + lm) * LDSU + bhalf];
            const uint4 d0 = ((const uint4*)p)[0];        // K [kb, kb+8)
            const uint4 d1 = ((const uint4*)(p + 4))[0];  // K [kb+8, kb+16)
            v8u vb = {d0.x, d0.y, d0.z, d0.w, d1.x, d1.y, d1.z, d1.w};
            bF[t] = __builtin_bit_cast(v16bf, vb);
        }
#pragma unroll
        for (int s = 0; s < 2; ++s)
#pragma unroll
            for (int t = 0; t < 4; ++t)
                acc[s][t] = __builtin_amdgcn_wmma_f32_16x16x32_bf16(
                    false, aF[s], false, bF[t], (short)0, acc[s][t], false, false);

        if (more) {
            fillA(buf ^ 1, (kt + 1) * KSTEP);
            asm volatile("s_wait_asynccnt 0x0" ::: "memory");
        }
        __syncthreads();
        buf ^= 1;
    }

    // epilogue: bias + relu, f32 out
#pragma unroll
    for (int s = 0; s < 2; ++s) {
#pragma unroll
        for (int t = 0; t < 4; ++t) {
            const int col  = Nblk + wn * 64 + t * 16 + lm;
            const int rowb = Mblk + wm * 32 + s * 16 + mhi;
            const float bb = bias[col];
#pragma unroll
            for (int r = 0; r < 8; ++r)
                out[(size_t)(rowb + r) * EMBED + col] = fmaxf(acc[s][t][r] + bb, 0.0f);
        }
    }
}

// =====================================================================
extern "C" void kernel_launch(void* const* d_in, const int* in_sizes, int n_in,
                              void* d_out, int out_size, void* d_ws, size_t ws_size,
                              hipStream_t stream)
{
    (void)in_sizes; (void)n_in; (void)out_size; (void)ws_size;
    const float* audio  = (const float*)d_in[0];
    const float* visual = (const float*)d_in[1];
    const float* Wa     = (const float*)d_in[2];
    const float* ba     = (const float*)d_in[3];
    const float* Wv     = (const float*)d_in[4];
    const float* bv     = (const float*)d_in[5];
    const float* Wi     = (const float*)d_in[6];
    const float* bi     = (const float*)d_in[7];
    const float* Wo     = (const float*)d_in[8];
    const float* bo     = (const float*)d_in[9];
    const float* Wf     = (const float*)d_in[10];
    const float* bf     = (const float*)d_in[11];
    float* out = (float*)d_out;

    char* ws = (char*)d_ws;
    float*          P    = (float*)(ws);
    float*          G1   = (float*)(ws + (size_t)4  * 1024 * 1024);
    float*          G2   = (float*)(ws + (size_t)8  * 1024 * 1024);
    unsigned short* Wcat = (unsigned short*)(ws + (size_t)12 * 1024 * 1024);
    float*          cvec = (float*)(ws + (size_t)20 * 1024 * 1024);
    float*          btot = (float*)(ws + (size_t)20 * 1024 * 1024 + 4096);

    const float* Wvin = Wi + (size_t)2 * EMBED * EMBED;   // Wi[2E:3E, :]

    // weight folding in f32 WMMA (tiny vs main GEMM)
    wgemm_f32<false><<<dim3(64, 64),  32, 0, stream>>>(Wo,         EMBED,     Wvin, EMBED,  P,    EMBED, EMBED);
    wgemm_f32<false><<<dim3(64, 64),  32, 0, stream>>>(Wf,         2 * EMBED, P,    EMBED,  G1,   EMBED, EMBED);
    wgemm_f32<false><<<dim3(64, 64),  32, 0, stream>>>(Wf + EMBED, 2 * EMBED, P,    EMBED,  G2,   EMBED, EMBED);
    wgemm_f32<true ><<<dim3(128, 64), 32, 0, stream>>>(G2,         EMBED,     Wa,   IN_DIM, Wcat,          KTOT, EMBED);
    wgemm_f32<true ><<<dim3(128, 64), 32, 0, stream>>>(G1,         EMBED,     Wv,   IN_DIM, Wcat + IN_DIM, KTOT, EMBED);
    bias_c_kernel  <<<4, 256, 0, stream>>>(Wo, bi, bo, cvec);
    bias_tot_kernel<<<4, 256, 0, stream>>>(Wf, bf, G1, G2, ba, bv, cvec, btot);

    // single HBM-bound fused GEMM over the whole batch
    fused_out<<<dim3(8, 256), 256, 0, stream>>>(audio, visual, Wcat, btot, out);
}
